// BiaffineParser_34187939676527
// MI455X (gfx1250) — compile-verified, matrix operand order checked by reading, not run on hardware
//
#include <hip/hip_runtime.h>

typedef __attribute__((ext_vector_type(16))) _Float16 v16h;
typedef __attribute__((ext_vector_type(8)))  float    v8f;

namespace {
constexpr int BATCH = 64;
constexpr int SEQL  = 128;
constexpr int ROWS  = BATCH * SEQL;   // 8192
constexpr int HID   = 400;
constexpr int G4    = 4 * HID;        // 1600
constexpr int HPAD  = 416;            // 400 padded to multiple of 32
constexpr int EPAD  = 128;            // 125 padded to 128
constexpr int H2    = 800;            // 2*HID
constexpr int DA    = 500, DAP = 512;
constexpr int DR    = 100, DRP = 128;
constexpr int NR    = 46;
constexpr int URN   = 4608;           // 46*100 padded to mult of 64
constexpr int CHUNK = 1024;
constexpr int NBLK  = 25 * 4 * 2;     // persistent LSTM grid (jtiles*mtiles*dirs)
}

// ---------------------------------------------------------------------------
// WMMA helpers (CDNA5 wave32; 16x16x32 f16 -> f32).
// Fragment: lane L holds row (tile + (L&15)); halves 0..7 = K[k+h .. k+h+7],
// 8..15 = K[k+16+h ..], h = (L>=16 ? 8 : 0). Same addressing for A (row=M)
// and B (row=N, weights stored (N,K)) => D = A * W^T.
// ---------------------------------------------------------------------------
__device__ __forceinline__ v16h load_frag(const _Float16* p) {
  union { v16h v; uint4 q[2]; } u;
  u.q[0] = *(const uint4*)(p);
  u.q[1] = *(const uint4*)(p + 16);
  return u.v;
}
__device__ __forceinline__ v8f wmma_f16(v16h a, v16h b, v8f c) {
  return __builtin_amdgcn_wmma_f32_16x16x32_f16(false, a, false, b, (short)0, c,
                                                false, false);
}
__device__ __forceinline__ float sigm(float x) { return 1.0f / (1.0f + __expf(-x)); }

// ---------------------------------------------------------------------------
// Generic WMMA GEMM: C(MxN) = A(MxK) * Bw(NxK)^T + bias, optional ReLU.
// M % 64 == 0, N % 64 == 0, K % 32 == 0. Columns n >= Nreal forced to 0.
// block = (32,4): 4 waves; each wave does a 64x16 strip (4 M-tiles) so one
// B fragment feeds 4 independent WMMAs (better L2 intensity + ILP).
// grid = (N/64, M/64).
// ---------------------------------------------------------------------------
__global__ __launch_bounds__(128) void gemm16(
    const _Float16* __restrict__ A, const _Float16* __restrict__ Bw,
    const float* __restrict__ bias, float* __restrict__ C,
    _Float16* __restrict__ Ch, int M, int N, int K, int Nreal, int relu) {
  const int lane  = threadIdx.x;
  const int tn    = blockIdx.x * 4 + threadIdx.y;
  const int m0    = blockIdx.y * 64;
  const int l15   = lane & 15;
  const int khalf = (lane >> 4) << 3;  // 0 or 8
  const _Float16* brow = Bw + (size_t)(tn * 16 + l15) * K + khalf;
  const _Float16* arow = A + (size_t)(m0 + l15) * K + khalf;
  const size_t a16 = (size_t)16 * K;
  v8f acc[4] = {{}, {}, {}, {}};
  for (int k = 0; k < K; k += 32) {
    v16h b = load_frag(brow + k);
#pragma unroll
    for (int i = 0; i < 4; ++i) {
      v16h a = load_frag(arow + (size_t)i * a16 + k);
      acc[i] = wmma_f16(a, b, acc[i]);
    }
  }
  const int n = tn * 16 + l15;
  float bv = (bias && n < Nreal) ? bias[n] : 0.0f;
#pragma unroll
  for (int i = 0; i < 4; ++i) {
    const int mbase = m0 + i * 16 + khalf;
#pragma unroll
    for (int r = 0; r < 8; ++r) {
      float v = acc[i][r] + bv;
      if (relu) v = fmaxf(v, 0.f);
      if (n >= Nreal) v = 0.f;
      size_t idx = (size_t)(mbase + r) * N + n;
      if (C)  C[idx]  = v;
      if (Ch) Ch[idx] = (_Float16)v;
    }
  }
}

// ---------------------------------------------------------------------------
// Persistent BiLSTM layer kernel. grid = (25 jtiles, 4 mtiles, 2 dirs),
// one wave per block, all 200 blocks co-resident.
//  - Wh gate-tile slice (4*16 rows x 416) staged to LDS once, then all 128
//    steps read B-fragments via ds_load_b128.
//  - Cell state c lives in registers (matches WMMA C layout, 8 f32/lane).
//  - h double-buffered in global (cross-block dependency), one device-wide
//    monotonic barrier per step (atomic counter + s_sleep spin).
// ---------------------------------------------------------------------------
__global__ __launch_bounds__(32) void lstm_layer(
    const float* __restrict__ preF, const float* __restrict__ preB,
    const _Float16* __restrict__ whF, const _Float16* __restrict__ whB,
    _Float16* __restrict__ hF0, _Float16* __restrict__ hF1,
    _Float16* __restrict__ hB0, _Float16* __restrict__ hB1,
    _Float16* __restrict__ y, unsigned* __restrict__ bar) {
  const int tj  = blockIdx.x;
  const int tm  = blockIdx.y;
  const int dir = blockIdx.z;
  const float*    pre = dir ? preB : preF;
  const _Float16* wh  = dir ? whB : whF;
  _Float16* h0 = dir ? hB0 : hF0;
  _Float16* h1 = dir ? hB1 : hF1;

  const int lane  = threadIdx.x;
  const int l15   = lane & 15;
  const int khalf = (lane >> 4) << 3;

  __shared__ _Float16 wlds[4 * 16 * HPAD];  // 53,248 B

  // Stage this block's Wh slice: gate g, tile row rr -> lds row g*16+rr.
  constexpr int QPR = HPAD / 8;  // uint4 (8 halfs) per row = 52
  for (int i = lane; i < 64 * QPR; i += 32) {
    int row = i / QPR, q = i % QPR;
    int g = row >> 4, rr = row & 15;
    const uint4* src =
        (const uint4*)(wh + (size_t)(g * HID + tj * 16 + rr) * HPAD) + q;
    *((uint4*)(wlds + (size_t)row * HPAD) + q) = *src;
  }
  // Single wave: compiler-inserted DScnt/LOADcnt waits order store->load.

  float creg[8] = {0.f, 0.f, 0.f, 0.f, 0.f, 0.f, 0.f, 0.f};
  const int n = tj * 16 + l15;

  for (int t = 0; t < SEQL; ++t) {
    const int teff = dir ? (SEQL - 1 - t) : t;
    const _Float16* hprev = (t & 1) ? h1 : h0;
    _Float16*       hnext = (t & 1) ? h0 : h1;
    const _Float16* arow = hprev + (size_t)(tm * 16 + l15) * HPAD + khalf;
    v8f acc[4] = {{}, {}, {}, {}};
    for (int k = 0; k < HPAD; k += 32) {
      v16h a = load_frag(arow + k);
#pragma unroll
      for (int g = 0; g < 4; ++g) {
        v16h b = load_frag(wlds + (size_t)(g * 16 + l15) * HPAD + khalf + k);
        acc[g] = wmma_f16(a, b, acc[g]);
      }
    }
#pragma unroll
    for (int r = 0; r < 8; ++r) {
      const int b = tm * 16 + khalf + r;  // batch index
      const size_t prow = ((size_t)b * SEQL + teff) * G4;
      float zi = pre[prow + 0 * HID + n] + acc[0][r];
      float zf = pre[prow + 1 * HID + n] + acc[1][r];
      float zg = pre[prow + 2 * HID + n] + acc[2][r];
      float zo = pre[prow + 3 * HID + n] + acc[3][r];
      float cn = sigm(zf) * creg[r] + sigm(zi) * tanhf(zg);
      float hn = sigm(zo) * tanhf(cn);
      creg[r] = cn;
      hnext[(size_t)b * HPAD + n] = (_Float16)hn;
      y[((size_t)b * SEQL + teff) * H2 + dir * HID + n] = (_Float16)hn;
    }
    // Device-wide step barrier (monotonic target, no reset needed in-kernel).
    __threadfence();
    if (lane == 0) {
      __hip_atomic_fetch_add(bar, 1u, __ATOMIC_ACQ_REL, __HIP_MEMORY_SCOPE_AGENT);
      const unsigned target = (unsigned)(t + 1) * (unsigned)NBLK;
      while (__hip_atomic_load(bar, __ATOMIC_ACQUIRE, __HIP_MEMORY_SCOPE_AGENT) <
             target)
        __builtin_amdgcn_s_sleep(1);
    }
    __threadfence();
  }
}

__global__ void lstm_init(_Float16* a, _Float16* b, _Float16* c, _Float16* d) {
  int i = blockIdx.x * blockDim.x + threadIdx.x;
  if (i < BATCH * HPAD) {
    a[i] = (_Float16)0.f; b[i] = (_Float16)0.f;
    c[i] = (_Float16)0.f; d[i] = (_Float16)0.f;
  }
}

__global__ void zero_u32(unsigned* p) {
  if (threadIdx.x == 0 && blockIdx.x == 0) *p = 0u;
}

// ---------------------------------------------------------------------------
// Batched biaffine: S[b,i,j] = T1[b,i,:] . Hah[b,j,:] + biasv[b,j]
// grid = (2, 2, 64), block = (32,4); each wave: 4 i-tiles x 1 j-tile, K=512.
// ---------------------------------------------------------------------------
__global__ __launch_bounds__(128) void biaffine_s(
    const _Float16* __restrict__ T1h, const _Float16* __restrict__ Hah,
    const float* __restrict__ biasv, float* __restrict__ S) {
  const int bb    = blockIdx.z;
  const int tn    = blockIdx.x * 4 + threadIdx.y;  // j tile 0..7
  const int m0    = blockIdx.y * 64;               // i base
  const int lane  = threadIdx.x;
  const int l15   = lane & 15;
  const int khalf = (lane >> 4) << 3;
  const _Float16* brow = Hah + ((size_t)bb * SEQL + tn * 16 + l15) * DAP + khalf;
  const _Float16* arow = T1h + ((size_t)bb * SEQL + m0 + l15) * DAP + khalf;
  const size_t a16 = (size_t)16 * DAP;
  v8f acc[4] = {{}, {}, {}, {}};
  for (int k = 0; k < DAP; k += 32) {
    v16h b = load_frag(brow + k);
#pragma unroll
    for (int i = 0; i < 4; ++i)
      acc[i] = wmma_f16(load_frag(arow + (size_t)i * a16 + k), b, acc[i]);
  }
  const int j    = tn * 16 + l15;
  const float bv = biasv[bb * SEQL + j];
#pragma unroll
  for (int i = 0; i < 4; ++i) {
    const int mbase = m0 + i * 16 + khalf;
#pragma unroll
    for (int r = 0; r < 8; ++r)
      S[(size_t)bb * SEQL * SEQL + (size_t)(mbase + r) * SEQL + j] =
          acc[i][r] + bv;
  }
}

// ------------------------- small VALU kernels ------------------------------
__global__ void embed_k(const int* __restrict__ words, const int* __restrict__ pos,
                        const float* __restrict__ wemb, const float* __restrict__ pemb,
                        _Float16* __restrict__ x) {
  size_t i = (size_t)blockIdx.x * blockDim.x + threadIdx.x;
  if (i >= (size_t)ROWS * EPAD) return;
  int row = (int)(i >> 7), col = (int)(i & 127);
  float v = 0.f;
  if (col < 100)      v = wemb[(size_t)words[row] * 100 + col];
  else if (col < 125) v = pemb[(size_t)pos[row] * 25 + (col - 100)];
  x[i] = (_Float16)v;
}

__global__ void cvt_pad(const float* __restrict__ src, _Float16* __restrict__ dst,
                        int N, int K, int Kp) {
  size_t i = (size_t)blockIdx.x * blockDim.x + threadIdx.x;
  if (i >= (size_t)N * Kp) return;
  int n = (int)(i / Kp), k = (int)(i % Kp);
  dst[i] = (k < K) ? (_Float16)src[(size_t)n * K + k] : (_Float16)0.f;
}

// dst[n][k] = src[k][n]  (src is Ksrc x Nsrc), zero-padded to Np x Kp
__global__ void cvt_pad_T(const float* __restrict__ src, _Float16* __restrict__ dst,
                          int Nsrc, int Ksrc, int Np, int Kp) {
  size_t i = (size_t)blockIdx.x * blockDim.x + threadIdx.x;
  if (i >= (size_t)Np * Kp) return;
  int n = (int)(i / Kp), k = (int)(i % Kp);
  dst[i] = (n < Nsrc && k < Ksrc) ? (_Float16)src[(size_t)k * Nsrc + n]
                                  : (_Float16)0.f;
}

// U_rel (100,46,100) -> U'[(r*100+k)][d] padded to (4608,128)
__global__ void cvt_urel(const float* __restrict__ U, _Float16* __restrict__ dst) {
  size_t i = (size_t)blockIdx.x * blockDim.x + threadIdx.x;
  if (i >= (size_t)URN * DRP) return;
  int n = (int)(i >> 7), d = (int)(i & 127);
  _Float16 v = (_Float16)0.f;
  if (n < NR * DR && d < DR) {
    int r = n / DR, k = n % DR;
    v = (_Float16)U[((size_t)d * NR + r) * DR + k];
  }
  dst[i] = v;
}

__global__ void dotrow_k(const float* __restrict__ Hah, const float* __restrict__ barc,
                         float* __restrict__ out) {
  int row = blockIdx.x * blockDim.x + threadIdx.x;
  if (row >= ROWS) return;
  const float* p = Hah + (size_t)row * DAP;
  float s = 0.f;
  for (int d = 0; d < DA; ++d) s += p[d] * barc[d];
  out[row] = s;
}

__global__ void argmax_k(const float* __restrict__ S, int* __restrict__ head_i,
                         float* __restrict__ head_f) {
  int row = blockIdx.x * blockDim.x + threadIdx.x;
  if (row >= ROWS) return;
  const float* p = S + (size_t)row * SEQL;
  float best = p[0];
  int bj = 0;
  for (int j = 1; j < SEQL; ++j) {
    float v = p[j];
    if (v > best) { best = v; bj = j; }
  }
  head_i[row] = bj;
  head_f[row] = (float)bj;
}

__global__ void gather_k(const _Float16* __restrict__ Hrh_h,
                         const int* __restrict__ head, _Float16* __restrict__ sel) {
  size_t i = (size_t)blockIdx.x * blockDim.x + threadIdx.x;
  if (i >= (size_t)ROWS * DRP) return;
  int row = (int)(i >> 7), k = (int)(i & 127);
  int bb = row >> 7;
  sel[i] = Hrh_h[(((size_t)bb << 7) + head[row]) * DRP + k];
}

// inter[row,r] = sum_k M1[rowlocal, r*100+k] * Hrd[row,k]
__global__ void rel_reduce(const float* __restrict__ M1, const float* __restrict__ Hrd,
                           float* __restrict__ inter, int rowbase) {
  int idx = blockIdx.x * blockDim.x + threadIdx.x;
  if (idx >= CHUNK * NR) return;
  int rl = idx / NR, r = idx % NR;
  int row = rowbase + rl;
  const float* m = M1 + (size_t)rl * URN + r * DR;
  const float* d = Hrd + (size_t)row * DRP;
  float s = 0.f;
  for (int k = 0; k < DR; ++k) s += m[k] * d[k];
  inter[(size_t)row * NR + r] = s;
}

__global__ void rel_final(const float* __restrict__ inter, const float* __restrict__ Hrh_f,
                          const float* __restrict__ Hrd_f, const int* __restrict__ head,
                          const float* __restrict__ Wrel, const float* __restrict__ brel,
                          float* __restrict__ outR) {
  int idx = blockIdx.x * blockDim.x + threadIdx.x;
  if (idx >= ROWS * NR) return;
  int row = idx / NR, r = idx % NR;
  int bb = row >> 7;
  const float* sel = Hrh_f + (((size_t)bb << 7) + head[row]) * DRP;
  const float* rd  = Hrd_f + (size_t)row * DRP;
  float s = brel[r];
  for (int d = 0; d < DR; ++d) s += (sel[d] + rd[d]) * Wrel[d * NR + r];
  outR[idx] = inter[idx] + s;
}

// ---------------------------------------------------------------------------
extern "C" void kernel_launch(void* const* d_in, const int* in_sizes, int n_in,
                              void* d_out, int out_size, void* d_ws, size_t ws_size,
                              hipStream_t stream) {
  (void)in_sizes; (void)n_in; (void)out_size; (void)ws_size;
  // Inputs in setup_inputs() dict order: params first, then words, pos.
  const float* word_emb = (const float*)d_in[0];
  const float* pos_emb  = (const float*)d_in[1];
  const float* wi[3][2] = {{(const float*)d_in[2],  (const float*)d_in[5]},
                           {(const float*)d_in[8],  (const float*)d_in[11]},
                           {(const float*)d_in[14], (const float*)d_in[17]}};
  const float* wh[3][2] = {{(const float*)d_in[3],  (const float*)d_in[6]},
                           {(const float*)d_in[9],  (const float*)d_in[12]},
                           {(const float*)d_in[15], (const float*)d_in[18]}};
  const float* bi[3][2] = {{(const float*)d_in[4],  (const float*)d_in[7]},
                           {(const float*)d_in[10], (const float*)d_in[13]},
                           {(const float*)d_in[16], (const float*)d_in[19]}};
  const float* Wad = (const float*)d_in[20]; const float* bad = (const float*)d_in[21];
  const float* Wah = (const float*)d_in[22]; const float* bah = (const float*)d_in[23];
  const float* Wrd = (const float*)d_in[24]; const float* brd = (const float*)d_in[25];
  const float* Wrh = (const float*)d_in[26]; const float* brh = (const float*)d_in[27];
  const float* Warc = (const float*)d_in[28]; const float* barc = (const float*)d_in[29];
  const float* Urel = (const float*)d_in[30];
  const float* Wrel = (const float*)d_in[31];
  const float* brel = (const float*)d_in[32];
  const int* words = (const int*)d_in[33];
  const int* pos   = (const int*)d_in[34];

  float* outS = (float*)d_out;                       // 64*128*128
  float* outR = outS + (size_t)BATCH * SEQL * SEQL;  // 64*128*46
  float* outH = outR + (size_t)ROWS * NR;            // 64*128

  // ---- workspace layout ----
  char* wsp = (char*)d_ws;
  size_t off = 0;
  auto alloc = [&](size_t bytes) -> void* {
    off = (off + 255) & ~(size_t)255;
    void* p = wsp + off;
    off += bytes;
    return p;
  };
  _Float16* x_h = (_Float16*)alloc((size_t)ROWS * EPAD * 2);
  _Float16* wi_h[3][2];
  _Float16* wh_h[3][2];
  const int kin[3] = {EPAD, H2, H2};
  for (int l = 0; l < 3; ++l)
    for (int d = 0; d < 2; ++d) {
      wi_h[l][d] = (_Float16*)alloc((size_t)G4 * kin[l] * 2);
      wh_h[l][d] = (_Float16*)alloc((size_t)G4 * HPAD * 2);
    }
  _Float16* wah_h  = (_Float16*)alloc((size_t)DAP * H2 * 2);
  _Float16* wad_h  = (_Float16*)alloc((size_t)DAP * H2 * 2);
  _Float16* wrh_h  = (_Float16*)alloc((size_t)DRP * H2 * 2);
  _Float16* wrd_h  = (_Float16*)alloc((size_t)DRP * H2 * 2);
  _Float16* warc_h = (_Float16*)alloc((size_t)DAP * DAP * 2);
  _Float16* urel_h = (_Float16*)alloc((size_t)URN * DRP * 2);
  float*    preF   = (float*)alloc((size_t)ROWS * G4 * 4);
  float*    preB   = (float*)alloc((size_t)ROWS * G4 * 4);
  _Float16* y0     = (_Float16*)alloc((size_t)ROWS * H2 * 2);
  _Float16* y1     = (_Float16*)alloc((size_t)ROWS * H2 * 2);
  _Float16* hF0    = (_Float16*)alloc((size_t)BATCH * HPAD * 2);
  _Float16* hF1    = (_Float16*)alloc((size_t)BATCH * HPAD * 2);
  _Float16* hB0    = (_Float16*)alloc((size_t)BATCH * HPAD * 2);
  _Float16* hB1    = (_Float16*)alloc((size_t)BATCH * HPAD * 2);
  float*    Hah_f  = (float*)alloc((size_t)ROWS * DAP * 4);
  _Float16* Hah_h  = (_Float16*)alloc((size_t)ROWS * DAP * 2);
  _Float16* Had_h  = (_Float16*)alloc((size_t)ROWS * DAP * 2);
  float*    Hrh_f  = (float*)alloc((size_t)ROWS * DRP * 4);
  _Float16* Hrh_h  = (_Float16*)alloc((size_t)ROWS * DRP * 2);
  float*    Hrd_f  = (float*)alloc((size_t)ROWS * DRP * 4);
  _Float16* T1_h   = (_Float16*)alloc((size_t)ROWS * DAP * 2);
  _Float16* sel_h  = (_Float16*)alloc((size_t)ROWS * DRP * 2);
  float*    biasv  = (float*)alloc((size_t)ROWS * 4);
  int*      head_i = (int*)alloc((size_t)ROWS * 4);
  unsigned* bar    = (unsigned*)alloc(256);
  float*    M1     = (float*)alloc((size_t)CHUNK * URN * 4);
  float*    inter  = (float*)alloc((size_t)ROWS * NR * 4);

  auto cdiv = [](size_t a, size_t b) { return (unsigned)((a + b - 1) / b); };

  // ---- weight conversion (f32 -> f16, padded / transposed for WMMA) ----
  for (int l = 0; l < 3; ++l)
    for (int d = 0; d < 2; ++d) {
      int ksrc = (l == 0) ? 125 : H2;
      cvt_pad<<<cdiv((size_t)G4 * kin[l], 256), 256, 0, stream>>>(
          wi[l][d], wi_h[l][d], G4, ksrc, kin[l]);
      cvt_pad<<<cdiv((size_t)G4 * HPAD, 256), 256, 0, stream>>>(
          wh[l][d], wh_h[l][d], G4, HID, HPAD);
    }
  cvt_pad_T<<<cdiv((size_t)DAP * H2, 256), 256, 0, stream>>>(Wah, wah_h, DA, H2, DAP, H2);
  cvt_pad_T<<<cdiv((size_t)DAP * H2, 256), 256, 0, stream>>>(Wad, wad_h, DA, H2, DAP, H2);
  cvt_pad_T<<<cdiv((size_t)DRP * H2, 256), 256, 0, stream>>>(Wrh, wrh_h, DR, H2, DRP, H2);
  cvt_pad_T<<<cdiv((size_t)DRP * H2, 256), 256, 0, stream>>>(Wrd, wrd_h, DR, H2, DRP, H2);
  cvt_pad_T<<<cdiv((size_t)DAP * DAP, 256), 256, 0, stream>>>(Warc, warc_h, DA, DA, DAP, DAP);
  cvt_urel<<<cdiv((size_t)URN * DRP, 256), 256, 0, stream>>>(Urel, urel_h);

  // ---- embedding gather ----
  embed_k<<<cdiv((size_t)ROWS * EPAD, 256), 256, 0, stream>>>(words, pos, word_emb,
                                                              pos_emb, x_h);

  // ---- 3 BiLSTM layers (persistent kernel per layer) ----
  const dim3 gblk(32, 4, 1);
  _Float16* lin  = x_h;
  _Float16* lout = y0;
  for (int l = 0; l < 3; ++l) {
    int K = kin[l];
    dim3 pg(G4 / 64, ROWS / 64, 1);
    gemm16<<<pg, gblk, 0, stream>>>(lin, wi_h[l][0], bi[l][0], preF, nullptr,
                                    ROWS, G4, K, G4, 0);
    gemm16<<<pg, gblk, 0, stream>>>(lin, wi_h[l][1], bi[l][1], preB, nullptr,
                                    ROWS, G4, K, G4, 0);
    lstm_init<<<cdiv((size_t)BATCH * HPAD, 256), 256, 0, stream>>>(hF0, hF1, hB0, hB1);
    zero_u32<<<1, 32, 0, stream>>>(bar);
    dim3 sg(HID / 16, BATCH / 16, 2);  // 25 x 4 x 2 = 200 blocks, co-resident
    lstm_layer<<<sg, 32, 0, stream>>>(preF, preB, wh_h[l][0], wh_h[l][1], hF0, hF1,
                                      hB0, hB1, lout, bar);
    lin  = lout;
    lout = (l == 0) ? y1 : y0;  // x -> y0 -> y1 -> y0(final)
  }
  _Float16* hfin = y0;  // final (8192, 800) f16 BiLSTM output

  // ---- MLP heads (ReLU) ----
  {
    dim3 g(DAP / 64, ROWS / 64, 1);
    gemm16<<<g, gblk, 0, stream>>>(hfin, wah_h, bah, Hah_f, Hah_h, ROWS, DAP, H2, DA, 1);
    gemm16<<<g, gblk, 0, stream>>>(hfin, wad_h, bad, nullptr, Had_h, ROWS, DAP, H2, DA, 1);
    dim3 g2(DRP / 64, ROWS / 64, 1);
    gemm16<<<g2, gblk, 0, stream>>>(hfin, wrh_h, brh, Hrh_f, Hrh_h, ROWS, DRP, H2, DR, 1);
    gemm16<<<g2, gblk, 0, stream>>>(hfin, wrd_h, brd, Hrd_f, nullptr, ROWS, DRP, H2, DR, 1);
  }

  // ---- biaffine arc scores ----
  {
    dim3 g(DAP / 64, ROWS / 64, 1);  // T1 = H_ad @ W_arc
    gemm16<<<g, gblk, 0, stream>>>(Had_h, warc_h, nullptr, nullptr, T1_h,
                                   ROWS, DAP, DAP, DAP, 0);
  }
  dotrow_k<<<cdiv(ROWS, 256), 256, 0, stream>>>(Hah_f, barc, biasv);
  {
    dim3 g(SEQL / 64, SEQL / 64, BATCH);
    biaffine_s<<<g, gblk, 0, stream>>>(T1_h, Hah_h, biasv, outS);
  }

  // ---- head prediction + gather ----
  argmax_k<<<cdiv(ROWS, 256), 256, 0, stream>>>(outS, head_i, outH);
  gather_k<<<cdiv((size_t)ROWS * DRP, 256), 256, 0, stream>>>(Hrh_h, head_i, sel_h);

  // ---- relation bilinear: M1 = sel @ U'^T (chunked), then contract with Hrd ----
  for (int c0 = 0; c0 < ROWS; c0 += CHUNK) {
    dim3 g(URN / 64, CHUNK / 64, 1);
    gemm16<<<g, gblk, 0, stream>>>(sel_h + (size_t)c0 * DRP, urel_h, nullptr, M1,
                                   nullptr, CHUNK, URN, DRP, URN, 0);
    rel_reduce<<<cdiv((size_t)CHUNK * NR, 256), 256, 0, stream>>>(M1, Hrd_f, inter, c0);
  }
  rel_final<<<cdiv((size_t)ROWS * NR, 256), 256, 0, stream>>>(
      inter, Hrh_f, Hrd_f, head_i, Wrel, brel, outR);
}
